// MambaVisionMixer_26766236189037
// MI455X (gfx1250) — matrix operand order repaired and checked by reading, not compiled, not run on hardware
//
#include <hip/hip_runtime.h>
#include <stdint.h>

// ---------------------------------------------------------------------------
// Mamba Vision Mixer for MI455X (gfx1250, wave32, WMMA).
//  * Big GEMMs: bf16 v_wmma_f32_16x16x32_bf16, 128x128 block tile, double-
//    buffered LDS fed by GLOBAL_LOAD_ASYNC_TO_LDS_B128 (ASYNCcnt) when the
//    toolchain exposes it (guarded, with synchronous fallback).
//  * Weights are pre-transposed to [N][K] bf16 once so both LDS tiles are
//    verbatim 16-byte row-major copies (no scatter transpose in the hot loop).
//  * Selective scan: 3-pass chunked linear recurrence (32 chunks of 64) to
//    break the L=2048 serialization.
// ---------------------------------------------------------------------------

typedef __attribute__((ext_vector_type(16))) __bf16        v16bf;
typedef __attribute__((ext_vector_type(8)))  float         v8f;
typedef __attribute__((ext_vector_type(4)))  unsigned int  u32x4;
typedef __attribute__((ext_vector_type(8)))  unsigned int  u32x8;
typedef __attribute__((ext_vector_type(4)))  int           i32x4;

#define D_MODEL   1024
#define D_INNER   2048
#define HALF      1024
#define DT_RANK   64
#define D_STATE   16
#define BATCH     8
#define SEQLEN    2048
#define MROWS     (BATCH * SEQLEN)          // 16384
#define XPROJ_N   (DT_RANK + 2 * D_STATE)   // 96
#define NCHUNK    32
#define CHUNK     (SEQLEN / NCHUNK)         // 64

#define CEILDIV(a, b) (((a) + (b) - 1) / (b))

// -------------------- CDNA5 async global->LDS copy path ---------------------
#if defined(__gfx1250__) && __has_builtin(__builtin_amdgcn_global_load_async_to_lds_b128)
#define HAVE_ASYNC_LDS 1
#else
#define HAVE_ASYNC_LDS 0
#endif

typedef __attribute__((address_space(1))) i32x4 gi32x4;   // global int4
typedef __attribute__((address_space(3))) i32x4 li32x4;   // LDS int4

__device__ __forceinline__ void copy16_g2lds(const unsigned short* g,
                                             unsigned short* l) {
#if HAVE_ASYNC_LDS
  __builtin_amdgcn_global_load_async_to_lds_b128(
      (gi32x4*)(uintptr_t)g, (li32x4*)(unsigned)(uintptr_t)l, 0, 0);
#else
  *reinterpret_cast<u32x4*>(l) = *reinterpret_cast<const u32x4*>(g);
#endif
}

__device__ __forceinline__ void async_tiles_wait() {
#if defined(__HIP_DEVICE_COMPILE__)
#if HAVE_ASYNC_LDS
#if __has_builtin(__builtin_amdgcn_s_wait_asynccnt)
  __builtin_amdgcn_s_wait_asynccnt(0);
#else
  asm volatile("s_wait_asynccnt 0x0" ::: "memory");
#endif
#endif
#endif
}

__device__ __forceinline__ unsigned short f2bf(float f) {
  unsigned int u = __float_as_uint(f);
  u += 0x7FFFu + ((u >> 16) & 1u);   // round-to-nearest-even truncation
  return (unsigned short)(u >> 16);
}
__device__ __forceinline__ float silu_f(float x) { return x / (1.f + __expf(-x)); }

// ---------------------------------------------------------------------------
// Elementwise fp32 -> bf16 cast (activations)
// ---------------------------------------------------------------------------
__global__ void cast_f32_bf16(const float* __restrict__ in,
                              unsigned short* __restrict__ out, int n) {
  int i = blockIdx.x * 256 + threadIdx.x;
  if (i < n) out[i] = f2bf(in[i]);
}

// ---------------------------------------------------------------------------
// fp32[R][C] -> bf16[C][R] transpose-cast (weights; makes GEMM-B K-major)
// ---------------------------------------------------------------------------
__global__ void cast_transpose_f32_bf16(const float* __restrict__ in,
                                        unsigned short* __restrict__ out,
                                        int R, int C) {
  int i = blockIdx.x * 256 + threadIdx.x;
  if (i >= R * C) return;
  int r = i / C, c = i % C;
  out[(size_t)c * R + r] = f2bf(in[i]);
}

// ---------------------------------------------------------------------------
// bf16 GEMM: C[M,N] = A[M,K] * Bt[N,K]^T (+bias), fp32 accumulate.
// Block 128x128, 8 wave32 (each 32x64 = 2x4 WMMA tiles), BK=32.
// Double-buffered LDS, async global->LDS fills.
// ---------------------------------------------------------------------------
#define BM 128
#define BN 128
#define BK 32
#define LDT_S 40   // bf16 elems per LDS tile row (32 + 8 pad), 80 B

__global__ __launch_bounds__(256) void gemm_bf16_wmma(
    float* __restrict__ C, const unsigned short* __restrict__ A,
    const unsigned short* __restrict__ Bt, int M, int N, int K,
    const float* __restrict__ bias) {
  __shared__ __align__(16) unsigned short As[2][BM * LDT_S];  // 2 x 10240 B
  __shared__ __align__(16) unsigned short Bs[2][BN * LDT_S];  // 2 x 10240 B

  const int tid   = threadIdx.x;
  const int wave  = tid >> 5;
  const int lane  = tid & 31;
  const int waveM = wave & 3;    // 4 waves along M, 32 rows each
  const int waveN = wave >> 2;   // 2 waves along N, 64 cols each
  const int blockM = blockIdx.y * BM;
  const int blockN = blockIdx.x * BN;

  const int row16 = lane & 15;
  const int half  = lane >> 4;

  v8f acc[2][4];
  const v8f vzero = {0.f, 0.f, 0.f, 0.f, 0.f, 0.f, 0.f, 0.f};
#pragma unroll
  for (int mi = 0; mi < 2; ++mi)
#pragma unroll
    for (int ni = 0; ni < 4; ++ni) acc[mi][ni] = vzero;

  const int nK = K / BK;

  // Per-thread fill assignment: 512 16-B chunks per tile, 2 per thread.
  // A tile rows are M; Bt tile rows are N; both are K-major in LDS.
  auto fill_tile = [&](int kt, int buf) {
    const int k0 = kt * BK;
#pragma unroll
    for (int i = 0; i < 2; ++i) {
      int idx = tid + i * 256;          // 0..511
      int r  = idx >> 2;                // tile row 0..127
      int ch = idx & 3;                 // 8-elem chunk along K
      copy16_g2lds(A + (size_t)(blockM + r) * K + k0 + ch * 8,
                   &As[buf][r * LDT_S + ch * 8]);
      int gcol = blockN + r;
      if (gcol >= N) gcol = N - 1;      // clamp (stores are guarded later)
      copy16_g2lds(Bt + (size_t)gcol * K + k0 + ch * 8,
                   &Bs[buf][r * LDT_S + ch * 8]);
    }
  };

  fill_tile(0, 0);
  async_tiles_wait();
  __syncthreads();

  for (int kt = 0; kt < nK; ++kt) {
    const int buf = kt & 1;
    if (kt + 1 < nK) fill_tile(kt + 1, buf ^ 1);

    // L2 warm for the tile after next (global_prefetch_b8)
    if (kt + 2 < nK) {
      __builtin_prefetch(A + (size_t)(blockM + (tid >> 1)) * K + (kt + 2) * BK,
                         0, 1);
    }

    // ---- A fragments: lanes 0-15 K{0..7,16..23}, 16-31 K{8..15,24..31} ----
    v16bf afrag[2];
#pragma unroll
    for (int mi = 0; mi < 2; ++mi) {
      int mrow = waveM * 32 + mi * 16 + row16;
      const unsigned short* base = &As[buf][mrow * LDT_S + half * 8];
      u32x4 lo = *reinterpret_cast<const u32x4*>(base);
      u32x4 hi = *reinterpret_cast<const u32x4*>(base + 16);
      u32x8 r;
      r[0] = lo[0]; r[1] = lo[1]; r[2] = lo[2]; r[3] = lo[3];
      r[4] = hi[0]; r[5] = hi[1]; r[6] = hi[2]; r[7] = hi[3];
      afrag[mi] = __builtin_bit_cast(v16bf, r);
    }

    // ---- B fragments: lane = column, lanes 0-15 K{0..7,16..23}, 16-31 K{8..15,24..31} ---
    v16bf bfrag[4];
#pragma unroll
    for (int ni = 0; ni < 4; ++ni) {
      int ncol = waveN * 64 + ni * 16 + row16;
      const unsigned short* base = &Bs[buf][ncol * LDT_S + half * 8];
      u32x4 lo = *reinterpret_cast<const u32x4*>(base);
      u32x4 hi = *reinterpret_cast<const u32x4*>(base + 16);
      u32x8 r;
      r[0] = lo[0]; r[1] = lo[1]; r[2] = lo[2]; r[3] = lo[3];
      r[4] = hi[0]; r[5] = hi[1]; r[6] = hi[2]; r[7] = hi[3];
      bfrag[ni] = __builtin_bit_cast(v16bf, r);
    }

    // ---- 8 WMMAs per K-step ----------------------------------------------
#pragma unroll
    for (int mi = 0; mi < 2; ++mi)
#pragma unroll
      for (int ni = 0; ni < 4; ++ni)
        acc[mi][ni] = __builtin_amdgcn_wmma_f32_16x16x32_bf16(
            false, afrag[mi], false, bfrag[ni], (short)0, acc[mi][ni],
            false, false);

    async_tiles_wait();   // tile kt+1 fully in LDS (per-wave ASYNCcnt)
    __syncthreads();      // all waves done with tile kt / fills visible
  }

  // ---- Epilogue: VGPR r holds M = r + 8*half, N = lane&15 ------------------
#pragma unroll
  for (int mi = 0; mi < 2; ++mi) {
#pragma unroll
    for (int ni = 0; ni < 4; ++ni) {
      int col = blockN + waveN * 64 + ni * 16 + row16;
      if (col < N) {
        float bv = bias ? bias[col] : 0.f;
#pragma unroll
        for (int r = 0; r < 8; ++r) {
          int rowg = blockM + waveM * 32 + mi * 16 + half * 8 + r;
          C[(size_t)rowg * N + col] = acc[mi][ni][r] + bv;
        }
      }
    }
  }
}

// ---------------------------------------------------------------------------
// Depthwise conv1d (W=4, SAME pad = (1,2)) + SiLU over one half of xz.
// sel=0: xh -> f32 u buffer + bf16 GEMM-A buffer
// sel=1: z  -> bf16 into yz[:,1024:2048]
// ---------------------------------------------------------------------------
__global__ void conv_silu(const float* __restrict__ xz,
                          const float* __restrict__ kern,
                          float* __restrict__ f32_out,
                          unsigned short* __restrict__ bf_out,
                          int outStride, int colOff, int sel) {
  int i = blockIdx.x * 256 + threadIdx.x;
  if (i >= MROWS * HALF) return;
  int row = i / HALF, c = i % HALF;
  int b = row / SEQLEN, l = row % SEQLEN;
  int col = sel * HALF + c;
  float acc = 0.f;
#pragma unroll
  for (int w = 0; w < 4; ++w) {
    int ls = l + w - 1;
    if (ls >= 0 && ls < SEQLEN)
      acc += kern[w * HALF + c] * xz[(size_t)(b * SEQLEN + ls) * D_INNER + col];
  }
  float s = silu_f(acc);
  if (f32_out) f32_out[(size_t)row * HALF + c] = s;
  bf_out[(size_t)row * outStride + colOff + c] = f2bf(s);
}

// ---------------------------------------------------------------------------
// Extract dt-rank columns of x_dbl as bf16 for the dt GEMM
// ---------------------------------------------------------------------------
__global__ void dtcast(const float* __restrict__ xdbl,
                       unsigned short* __restrict__ dtp) {
  int i = blockIdx.x * 256 + threadIdx.x;
  if (i >= MROWS * DT_RANK) return;
  int row = i / DT_RANK, c = i % DT_RANK;
  dtp[i] = f2bf(xdbl[(size_t)row * XPROJ_N + c]);
}

// ---------------------------------------------------------------------------
// delta = softplus(dt_raw + 2*b_dt)   (bias double-added, faithful to ref)
// ---------------------------------------------------------------------------
__global__ void delta_softplus(const float* __restrict__ dtraw,
                               const float* __restrict__ b_dt,
                               float* __restrict__ delta) {
  int i = blockIdx.x * 256 + threadIdx.x;
  if (i >= MROWS * HALF) return;
  int c = i % HALF;
  float v = dtraw[i] + 2.f * b_dt[c];
  delta[i] = (v > 20.f) ? v : __logf(1.f + __expf(v));
}

// ---------------------------------------------------------------------------
// Chunked selective scan, pass A: per-chunk (prod dA, local h from h0=0)
// grid = (HALF/256, NCHUNK, BATCH)
// ---------------------------------------------------------------------------
__global__ void scan_chunkA(const float* __restrict__ delta,
                            const float* __restrict__ u,
                            const float* __restrict__ xdbl,
                            const float* __restrict__ A_log,
                            float* __restrict__ P, float* __restrict__ Hloc) {
  int d = blockIdx.x * 256 + threadIdx.x;
  int c = blockIdx.y, b = blockIdx.z;
  __shared__ float Bs[D_STATE];
  float negA[D_STATE], h[D_STATE], p[D_STATE];
#pragma unroll
  for (int n = 0; n < D_STATE; ++n) {
    negA[n] = -__expf(A_log[d * D_STATE + n]);
    h[n] = 0.f;
    p[n] = 1.f;
  }
  for (int l = c * CHUNK; l < (c + 1) * CHUNK; ++l) {
    size_t row = (size_t)b * SEQLEN + l;
    if (threadIdx.x < D_STATE)
      Bs[threadIdx.x] = xdbl[row * XPROJ_N + DT_RANK + threadIdx.x];
    __syncthreads();
    float dt = delta[row * HALF + d];
    float ut = u[row * HALF + d];
    float dbu = dt * ut;
#pragma unroll
    for (int n = 0; n < D_STATE; ++n) {
      float da = __expf(dt * negA[n]);
      h[n] = da * h[n] + dbu * Bs[n];
      p[n] *= da;
    }
    __syncthreads();
  }
  size_t o = ((size_t)(b * NCHUNK + c) * HALF + d) * D_STATE;
#pragma unroll
  for (int n = 0; n < D_STATE; ++n) {
    P[o + n] = p[n];
    Hloc[o + n] = h[n];
  }
}

// ---------------------------------------------------------------------------
// Pass B: tiny sequential combine over 32 chunk summaries per (b,d,n)
// ---------------------------------------------------------------------------
__global__ void scan_combine(const float* __restrict__ P,
                             const float* __restrict__ Hloc,
                             float* __restrict__ hstart) {
  int i = blockIdx.x * 256 + threadIdx.x;
  if (i >= BATCH * HALF * D_STATE) return;
  int n = i % D_STATE;
  int d = (i / D_STATE) % HALF;
  int b = i / (D_STATE * HALF);
  float h = 0.f;
  for (int c = 0; c < NCHUNK; ++c) {
    size_t o = ((size_t)(b * NCHUNK + c) * HALF + d) * D_STATE + n;
    hstart[o] = h;
    h = P[o] * h + Hloc[o];
  }
}

// ---------------------------------------------------------------------------
// Pass C: replay each chunk from its true initial state, emit y (+u*D) as
// bf16 into yz[:, 0:1024].  grid = (HALF/256, NCHUNK, BATCH)
// ---------------------------------------------------------------------------
__global__ void scan_chunkC(const float* __restrict__ delta,
                            const float* __restrict__ u,
                            const float* __restrict__ xdbl,
                            const float* __restrict__ A_log,
                            const float* __restrict__ Dvec,
                            const float* __restrict__ hstart,
                            unsigned short* __restrict__ yz) {
  int d = blockIdx.x * 256 + threadIdx.x;
  int c = blockIdx.y, b = blockIdx.z;
  __shared__ float Bs[D_STATE];
  __shared__ float Cs[D_STATE];
  float negA[D_STATE], h[D_STATE];
  size_t o = ((size_t)(b * NCHUNK + c) * HALF + d) * D_STATE;
#pragma unroll
  for (int n = 0; n < D_STATE; ++n) {
    negA[n] = -__expf(A_log[d * D_STATE + n]);
    h[n] = hstart[o + n];
  }
  float Dd = Dvec[d];
  for (int l = c * CHUNK; l < (c + 1) * CHUNK; ++l) {
    size_t row = (size_t)b * SEQLEN + l;
    if (threadIdx.x < D_STATE)
      Bs[threadIdx.x] = xdbl[row * XPROJ_N + DT_RANK + threadIdx.x];
    else if (threadIdx.x < 2 * D_STATE)
      Cs[threadIdx.x - D_STATE] =
          xdbl[row * XPROJ_N + DT_RANK + threadIdx.x];
    __syncthreads();
    float dt = delta[row * HALF + d];
    float ut = u[row * HALF + d];
    float dbu = dt * ut;
    float y = 0.f;
#pragma unroll
    for (int n = 0; n < D_STATE; ++n) {
      float da = __expf(dt * negA[n]);
      h[n] = da * h[n] + dbu * Bs[n];
      y += h[n] * Cs[n];
    }
    y += ut * Dd;
    yz[row * D_INNER + d] = f2bf(y);
    __syncthreads();
  }
}

// ---------------------------------------------------------------------------
// Host-side orchestration
// ---------------------------------------------------------------------------
extern "C" void kernel_launch(void* const* d_in, const int* in_sizes, int n_in,
                              void* d_out, int out_size, void* d_ws,
                              size_t ws_size, hipStream_t stream) {
  (void)in_sizes; (void)n_in; (void)out_size; (void)ws_size;

  const float* x      = (const float*)d_in[0];
  const float* W_in   = (const float*)d_in[1];
  const float* conv_x = (const float*)d_in[2];
  const float* conv_z = (const float*)d_in[3];
  const float* W_xp   = (const float*)d_in[4];
  const float* W_dt   = (const float*)d_in[5];
  const float* b_dt   = (const float*)d_in[6];
  const float* A_log  = (const float*)d_in[7];
  const float* Dvec   = (const float*)d_in[8];
  const float* W_out  = (const float*)d_in[9];
  const float* b_out  = (const float*)d_in[10];
  float* out = (float*)d_out;

  char* ws = (char*)d_ws;
  size_t off = 0;
  auto alloc = [&](size_t bytes) -> char* {
    char* p = ws + off;
    off += (bytes + 255) & ~(size_t)255;
    return p;
  };

  unsigned short* x_bf    = (unsigned short*)alloc((size_t)MROWS * D_MODEL * 2);
  // Weights stored pre-transposed: [N][K] bf16
  unsigned short* WinT  = (unsigned short*)alloc((size_t)D_INNER * D_MODEL * 2);
  unsigned short* WxpT  = (unsigned short*)alloc((size_t)XPROJ_N * HALF * 2);
  unsigned short* WdtT  = (unsigned short*)alloc((size_t)HALF * DT_RANK * 2);
  unsigned short* WoutT = (unsigned short*)alloc((size_t)D_MODEL * D_INNER * 2);
  float* xz    = (float*)alloc((size_t)MROWS * D_INNER * 4);
  float* u     = (float*)alloc((size_t)MROWS * HALF * 4);
  unsigned short* xh_bf = (unsigned short*)alloc((size_t)MROWS * HALF * 2);
  float* xdbl  = (float*)alloc((size_t)MROWS * XPROJ_N * 4);
  unsigned short* dtp_bf = (unsigned short*)alloc((size_t)MROWS * DT_RANK * 2);
  float* dtraw = (float*)alloc((size_t)MROWS * HALF * 4);
  float* delta = (float*)alloc((size_t)MROWS * HALF * 4);
  unsigned short* yz_bf = (unsigned short*)alloc((size_t)MROWS * D_INNER * 2);
  size_t ssz = (size_t)BATCH * NCHUNK * HALF * D_STATE * 4;
  float* Pbuf   = (float*)alloc(ssz);
  float* Hloc   = (float*)alloc(ssz);
  float* hstart = (float*)alloc(ssz);

  dim3 blk(256);
  auto gemm = [&](float* C, const unsigned short* Am, const unsigned short* Bm,
                  int M, int N, int K, const float* bias) {
    dim3 grid(CEILDIV(N, BN), CEILDIV(M, BM));
    gemm_bf16_wmma<<<grid, blk, 0, stream>>>(C, Am, Bm, M, N, K, bias);
  };

  // 1) casts: activations straight, weights transposed to [N][K]
  cast_f32_bf16<<<CEILDIV(MROWS * D_MODEL, 256), blk, 0, stream>>>(
      x, x_bf, MROWS * D_MODEL);
  cast_transpose_f32_bf16<<<CEILDIV(D_MODEL * D_INNER, 256), blk, 0, stream>>>(
      W_in, WinT, D_MODEL, D_INNER);
  cast_transpose_f32_bf16<<<CEILDIV(HALF * XPROJ_N, 256), blk, 0, stream>>>(
      W_xp, WxpT, HALF, XPROJ_N);
  cast_transpose_f32_bf16<<<CEILDIV(DT_RANK * HALF, 256), blk, 0, stream>>>(
      W_dt, WdtT, DT_RANK, HALF);
  cast_transpose_f32_bf16<<<CEILDIV(D_INNER * D_MODEL, 256), blk, 0, stream>>>(
      W_out, WoutT, D_INNER, D_MODEL);

  // 2) xz = x @ W_in
  gemm(xz, x_bf, WinT, MROWS, D_INNER, D_MODEL, nullptr);

  // 3) depthwise conv + SiLU on both halves
  conv_silu<<<CEILDIV(MROWS * HALF, 256), blk, 0, stream>>>(
      xz, conv_x, u, xh_bf, HALF, 0, 0);
  conv_silu<<<CEILDIV(MROWS * HALF, 256), blk, 0, stream>>>(
      xz, conv_z, nullptr, yz_bf, D_INNER, HALF, 1);

  // 4) x_dbl = xh @ W_xproj
  gemm(xdbl, xh_bf, WxpT, MROWS, XPROJ_N, HALF, nullptr);

  // 5) dt_raw = x_dbl[:, :64] @ W_dt ; delta = softplus(dt_raw + 2*b_dt)
  dtcast<<<CEILDIV(MROWS * DT_RANK, 256), blk, 0, stream>>>(xdbl, dtp_bf);
  gemm(dtraw, dtp_bf, WdtT, MROWS, HALF, DT_RANK, nullptr);
  delta_softplus<<<CEILDIV(MROWS * HALF, 256), blk, 0, stream>>>(
      dtraw, b_dt, delta);

  // 6) chunked selective scan -> yz[:, :1024]
  dim3 gscan(HALF / 256, NCHUNK, BATCH);
  scan_chunkA<<<gscan, blk, 0, stream>>>(delta, u, xdbl, A_log, Pbuf, Hloc);
  scan_combine<<<CEILDIV(BATCH * HALF * D_STATE, 256), blk, 0, stream>>>(
      Pbuf, Hloc, hstart);
  scan_chunkC<<<gscan, blk, 0, stream>>>(delta, u, xdbl, A_log, Dvec, hstart,
                                         yz_bf);

  // 7) out = [y, z] @ W_out + b_out
  gemm(out, yz_bf, WoutT, MROWS, D_MODEL, D_INNER, b_out);
}
